// WindowAttention_18640158064973
// MI455X (gfx1250) — compile-verified, matrix operand order checked by reading, not run on hardware
//
#include <hip/hip_runtime.h>

// ---------------------------------------------------------------------------
// Swin window attention for MI455X (gfx1250), bf16 WMMA w/ fp32 accumulate.
//   shapes fixed by reference: b=32, 56x56, C=384, 7x7 windows, 12 heads, hd=32
//   shift = 3 (56 > 7), pad = 0 (56 % 7 == 0)
// Pipeline:
//   K0: one-time fp32 -> bf16 weight conversion (qkv_w, proj_w)
//   K1: per-window shifted gather -> LDS (bf16, zero-padded) -> QKV GEMM
//   K2: per (window, head) attention: q@k^T -> softmax(+bias+mask) -> P@V
//   K3: per-window proj GEMM; A staged via global_load_async_to_lds_b128
// ---------------------------------------------------------------------------

typedef __attribute__((ext_vector_type(16))) __bf16 v16bf;
typedef __attribute__((ext_vector_type(8)))  float  v8f;
typedef __attribute__((ext_vector_type(4)))  float  v4f;

namespace {
constexpr int   kWH     = 7;
constexpr int   kN      = 49;          // window tokens
constexpr int   kNH     = 12;          // heads
constexpr int   kC      = 384;
constexpr int   kHD     = 32;          // head dim
constexpr int   kGrid   = 56;
constexpr int   kShift  = 3;
constexpr int   kBatch  = 32;
constexpr int   kNWin   = 64;          // 8x8 windows per image
constexpr int   kBW     = kBatch * kNWin;       // 2048 windows total
constexpr float kQScale = 0.17677669529663687f; // 1/sqrt(32)

constexpr size_t QSZ  = (size_t)kBW * kNH * kN * kHD;  // q (and k) elems
constexpr size_t VSZ  = (size_t)kBW * kNH * kHD * 64;  // v transposed, K pad 64
constexpr size_t AOSZ = (size_t)kBW * kN * kC;         // attention output
constexpr size_t W1SZ = (size_t)3 * kC * kC;           // qkv_w bf16 (1152x384)
constexpr size_t W2SZ = (size_t)kC * kC;               // proj_w bf16 (384x384)
} // namespace

__device__ inline v8f wmma_bf16(v16bf a, v16bf b, v8f c) {
  // D = A(16x32 bf16) * B(32x16 bf16) + C(16x16 f32)
  return __builtin_amdgcn_wmma_f32_16x16x32_bf16(false, a, false, b, (short)0, c,
                                                 false, false);
}

__device__ inline v8f zero_acc() {
  v8f z;
#pragma unroll
  for (int i = 0; i < 8; ++i) z[i] = 0.0f;
  return z;
}

// Per-lane 16-bit matrix fragment (A row / B column): element e maps to
// k = (e>=8 ? 16 : 0) + half*8 + (e%8)  -> two contiguous 8-elem chunks.
__device__ inline v16bf frag_bf16(const __bf16* row, int half) {
  v16bf f;
  const __bf16* p0 = row + half * 8;
  const __bf16* p1 = row + 16 + half * 8;
#pragma unroll
  for (int i = 0; i < 8; ++i) {
    f[i]     = p0[i];
    f[i + 8] = p1[i];
  }
  return f;
}

// ---------------------------------------------------------------------------
// Kernel 0: one-time fp32 -> bf16 weight conversion (4 elems / thread).
// ---------------------------------------------------------------------------
__global__ __launch_bounds__(256) void convert_w_kernel(
    const float* __restrict__ src, __bf16* __restrict__ dst, int n4) {
  int i = blockIdx.x * 256 + threadIdx.x;
  if (i < n4) {
    v4f f = *(const v4f*)(src + (size_t)i * 4);
#pragma unroll
    for (int j = 0; j < 4; ++j) dst[(size_t)i * 4 + j] = (__bf16)f[j];
  }
}

// ---------------------------------------------------------------------------
// Kernel 1: one block (256 thr = 8 waves) per window.
//   Stage gathered+shifted window rows into LDS as bf16 [64][384] (rows 49..63
//   zeroed), then 9 col-iterations x 8 waves cover all 1152 QKV columns.
//   q scaled, k as [wi][head][49][32]; v transposed+padded [wi][head][32][64].
// ---------------------------------------------------------------------------
__global__ __launch_bounds__(256) void swin_qkv_kernel(
    const float* __restrict__ x, const __bf16* __restrict__ qkv_wb,
    const float* __restrict__ qkv_b, __bf16* __restrict__ qws,
    __bf16* __restrict__ kws, __bf16* __restrict__ vws) {
  __shared__ __attribute__((aligned(16))) __bf16 Xs[64 * kC];  // 48 KB

  const int wi   = blockIdx.x;
  const int tid  = threadIdx.x;
  const int wave = tid >> 5;
  const int lane = tid & 31;
  const int half = lane >> 4;
  const int lmod = lane & 15;
  const int batch = wi >> 6;
  const int widx  = wi & 63;
  const int wh0 = widx >> 3, ww0 = widx & 7;

  // ---- stage window into LDS (fp32 gather -> bf16), zero-pad rows 49..63 ----
  for (int idx = tid; idx < 64 * (kC / 4); idx += 256) {
    int row = idx / (kC / 4);
    int g   = idx % (kC / 4);
    __bf16 o0, o1, o2, o3;
    if (row < kN) {
      int r = row / kWH, cc = row % kWH;
      int hh = (wh0 * kWH + r + kShift) % kGrid;   // roll by -shift folded in
      int ww = (ww0 * kWH + cc + kShift) % kGrid;
      const float* p =
          x + (((size_t)batch * kGrid + hh) * kGrid + ww) * kC + g * 4;
      v4f f = *(const v4f*)p;
      o0 = (__bf16)f[0]; o1 = (__bf16)f[1]; o2 = (__bf16)f[2]; o3 = (__bf16)f[3];
    } else {
      o0 = o1 = o2 = o3 = (__bf16)0.0f;
    }
    __bf16* d = &Xs[row * kC + g * 4];
    d[0] = o0; d[1] = o1; d[2] = o2; d[3] = o3;
  }
  __syncthreads();

  // ---- GEMM: M = 64 (padded), K = 384, N = 1152 (9 iters x 8 waves x 16) ----
  for (int it = 0; it < 9; ++it) {
    const int n0 = (it * 8 + wave) * 16;
    const __bf16* brow = qkv_wb + (size_t)(n0 + lmod) * kC;

    v8f acc[4];
#pragma unroll
    for (int mt = 0; mt < 4; ++mt) acc[mt] = zero_acc();

    for (int k0 = 0; k0 < kC; k0 += 32) {
      v16bf bf = frag_bf16(brow + k0, half);
#pragma unroll
      for (int mt = 0; mt < 4; ++mt) {
        v16bf af = frag_bf16(&Xs[(mt * 16 + lmod) * kC + k0], half);
        acc[mt] = wmma_bf16(af, bf, acc[mt]);
      }
    }

    const int   n     = n0 + lmod;
    const float bias  = qkv_b[n];
    const int   which = n / kC;          // 0=q 1=k 2=v
    const int   head  = (n % kC) / kHD;
    const int   d     = n % kHD;
    const size_t hb   = (size_t)wi * kNH + head;

#pragma unroll
    for (int mt = 0; mt < 4; ++mt) {
#pragma unroll
      for (int r = 0; r < 8; ++r) {
        int   m = mt * 16 + r + 8 * half;
        float v = acc[mt][r] + bias;
        if (which == 0) {
          if (m < kN) qws[(hb * kN + m) * kHD + d] = (__bf16)(v * kQScale);
        } else if (which == 1) {
          if (m < kN) kws[(hb * kN + m) * kHD + d] = (__bf16)v;
        } else {  // v transposed [d][64], zero-pad rows 49..63
          vws[(hb * kHD + d) * 64 + m] = (m < kN) ? (__bf16)v : (__bf16)0.0f;
        }
      }
    }
  }
}

// ---------------------------------------------------------------------------
// Kernel 2: per (window, head) attention. One wave per block.
//   S = q@k^T (16 WMMAs, K=32) -> LDS, softmax with analytic rel-pos bias and
//   shift mask, P bf16 -> LDS, out = P@V (16 WMMAs, K=64 in 2 steps).
// ---------------------------------------------------------------------------
__global__ __launch_bounds__(32) void swin_attn_kernel(
    const __bf16* __restrict__ qws, const __bf16* __restrict__ kws,
    const __bf16* __restrict__ vws, const float* __restrict__ rpb,
    __bf16* __restrict__ ao) {
  __shared__ float  Sb[64 * 64];
  __shared__ __bf16 Pb[64 * 64];

  const int blk  = blockIdx.x;
  const int wi   = blk / kNH;
  const int head = blk % kNH;
  const int lane = threadIdx.x & 31;
  const int half = lane >> 4;
  const int lmod = lane & 15;
  const int widx = wi & 63;
  const int wh0 = widx >> 3, ww0 = widx & 7;

  const size_t hb = (size_t)wi * kNH + head;
  const __bf16* qb = qws + hb * (kN * kHD);
  const __bf16* kb = kws + hb * (kN * kHD);
  const __bf16* vb = vws + hb * (kHD * 64);

  // ---- S = q @ k^T  (rows/cols padded to 64; K = 32, single WMMA step) ----
  v16bf qa[4];
#pragma unroll
  for (int mt = 0; mt < 4; ++mt)
    qa[mt] = frag_bf16(qb + (size_t)(mt * 16 + lmod) * kHD, half);

#pragma unroll
  for (int nt = 0; nt < 4; ++nt) {
    v16bf kf = frag_bf16(kb + (size_t)(nt * 16 + lmod) * kHD, half);
#pragma unroll
    for (int mt = 0; mt < 4; ++mt) {
      v8f dacc = wmma_bf16(qa[mt], kf, zero_acc());
#pragma unroll
      for (int r = 0; r < 8; ++r)
        Sb[(mt * 16 + r + 8 * half) * 64 + nt * 16 + lmod] = dacc[r];
    }
  }
  __syncthreads();

  // ---- softmax rows (each lane: rows lane, lane+32) ----
#pragma unroll
  for (int rr = 0; rr < 2; ++rr) {
    int i = lane + rr * 32;
    if (i < kN) {
      int ih = i / kWH, iw = i % kWH;
      // shift-mask region code; only boundary windows (index 7) mix regions
      int ci = ((wh0 == 7) ? (ih < 4 ? 1 : 2) : 0) * 3 +
               ((ww0 == 7) ? (iw < 4 ? 1 : 2) : 0);
      float mx = -3.0e38f;
      for (int j = 0; j < kN; ++j) {
        int jh = j / kWH, jw = j % kWH;
        int cj = ((wh0 == 7) ? (jh < 4 ? 1 : 2) : 0) * 3 +
                 ((ww0 == 7) ? (jw < 4 ? 1 : 2) : 0);
        float bias = rpb[((ih - jh + 6) * 13 + (iw - jw + 6)) * kNH + head];
        float s = Sb[i * 64 + j] + bias + ((ci != cj) ? -100.0f : 0.0f);
        Sb[i * 64 + j] = s;
        mx = fmaxf(mx, s);
      }
      float sum = 0.0f;
      for (int j = 0; j < kN; ++j) {
        float e = __expf(Sb[i * 64 + j] - mx);
        sum += e;
        Sb[i * 64 + j] = e;
      }
      float inv = 1.0f / sum;
      for (int j = 0; j < kN; ++j) Pb[i * 64 + j] = (__bf16)(Sb[i * 64 + j] * inv);
      for (int j = kN; j < 64; ++j) Pb[i * 64 + j] = (__bf16)0.0f;
    }
  }
  __syncthreads();

  // ---- out = P @ v  (K = 64, two k-steps; v is [d][64] so B frags are rows)
  v16bf pa[4][2];
#pragma unroll
  for (int mt = 0; mt < 4; ++mt)
#pragma unroll
    for (int ks = 0; ks < 2; ++ks)
      pa[mt][ks] = frag_bf16(&Pb[(mt * 16 + lmod) * 64 + ks * 32], half);

#pragma unroll
  for (int nt = 0; nt < 2; ++nt) {
    v16bf vf0 = frag_bf16(vb + (size_t)(nt * 16 + lmod) * 64 + 0, half);
    v16bf vf1 = frag_bf16(vb + (size_t)(nt * 16 + lmod) * 64 + 32, half);
#pragma unroll
    for (int mt = 0; mt < 4; ++mt) {
      v8f oacc = wmma_bf16(pa[mt][0], vf0, zero_acc());
      oacc     = wmma_bf16(pa[mt][1], vf1, oacc);
#pragma unroll
      for (int r = 0; r < 8; ++r) {
        int m = mt * 16 + r + 8 * half;
        if (m < kN)
          ao[((size_t)wi * kN + m) * kC + head * kHD + nt * 16 + lmod] =
              (__bf16)oacc[r];
      }
    }
  }
}

// ---------------------------------------------------------------------------
// Kernel 3: one block (256 thr = 8 waves) per window.
//   A (= attention output window, already bf16, fully contiguous 37632 B) is
//   staged into LDS via the CDNA5 async path: global_load_async_to_lds_b128
//   + s_wait_asynccnt. Rows 49..63 zero-padded. Then proj GEMM (K=384,
//   N=384 in 3 iters x 8 waves x 16) and fused window-reverse + roll(+3)
//   scatter of fp32 results.
// ---------------------------------------------------------------------------
__global__ __launch_bounds__(256) void swin_proj_kernel(
    const __bf16* __restrict__ ao, const __bf16* __restrict__ proj_wb,
    const float* __restrict__ proj_b, float* __restrict__ out) {
  __shared__ __attribute__((aligned(16))) __bf16 As[64 * kC];  // 48 KB

  const int wi   = blockIdx.x;
  const int tid  = threadIdx.x;
  const int wave = tid >> 5;
  const int lane = tid & 31;
  const int half = lane >> 4;
  const int lmod = lane & 15;
  const int batch = wi >> 6;
  const int widx  = wi & 63;
  const int wh0 = widx >> 3, ww0 = widx & 7;

  // ---- async-stage the 49x384 bf16 window (2352 x 16B, exact) ----
  const __bf16* abase = ao + (size_t)wi * kN * kC;
  const unsigned int lds_base = (unsigned int)(uintptr_t)(&As[0]);
  for (int idx = tid; idx < (kN * kC * 2) / 16; idx += 256) {
    unsigned long long g =
        (unsigned long long)(uintptr_t)abase + (unsigned long long)idx * 16u;
    unsigned int l = lds_base + (unsigned int)idx * 16u;
    asm volatile("global_load_async_to_lds_b128 %0, %1, off"
                 :: "v"(l), "v"(g) : "memory");
  }
  // zero-pad rows 49..63 (byte offset 37632, 8B-aligned stores)
  for (int idx = tid; idx < (64 - kN) * kC / 4; idx += 256) {
    *(unsigned long long*)&As[kN * kC + idx * 4] = 0ull;
  }
  asm volatile("s_wait_asynccnt 0x0" ::: "memory");
  __syncthreads();

  // ---- GEMM: M = 64 (padded), K = 384, N = 384 (3 iters x 8 waves x 16) ----
  for (int it = 0; it < 3; ++it) {
    const int n0 = (it * 8 + wave) * 16;
    const __bf16* brow = proj_wb + (size_t)(n0 + lmod) * kC;

    v8f acc[4];
#pragma unroll
    for (int mt = 0; mt < 4; ++mt) acc[mt] = zero_acc();

    for (int k0 = 0; k0 < kC; k0 += 32) {
      v16bf bf = frag_bf16(brow + k0, half);
#pragma unroll
      for (int mt = 0; mt < 4; ++mt) {
        v16bf af = frag_bf16(&As[(mt * 16 + lmod) * kC + k0], half);
        acc[mt] = wmma_bf16(af, bf, acc[mt]);
      }
    }

    const int   n    = n0 + lmod;
    const float bias = proj_b[n];
#pragma unroll
    for (int mt = 0; mt < 4; ++mt) {
#pragma unroll
      for (int r = 0; r < 8; ++r) {
        int m = mt * 16 + r + 8 * half;
        if (m < kN) {
          int rr = m / kWH, cc = m % kWH;
          int hh = (wh0 * kWH + rr + kShift) % kGrid;  // reverse + roll +3
          int ww = (ww0 * kWH + cc + kShift) % kGrid;
          out[(((size_t)batch * kGrid + hh) * kGrid + ww) * kC + n] =
              acc[mt][r] + bias;
        }
      }
    }
  }
}

// ---------------------------------------------------------------------------
extern "C" void kernel_launch(void* const* d_in, const int* in_sizes, int n_in,
                              void* d_out, int out_size, void* d_ws,
                              size_t ws_size, hipStream_t stream) {
  (void)in_sizes; (void)n_in; (void)out_size; (void)ws_size;
  const float* x      = (const float*)d_in[0];
  const float* qkv_w  = (const float*)d_in[1];
  const float* qkv_b  = (const float*)d_in[2];
  const float* proj_w = (const float*)d_in[3];
  const float* proj_b = (const float*)d_in[4];
  const float* rpb    = (const float*)d_in[5];
  float*       out    = (float*)d_out;

  // workspace layout (bf16): q | k | v_t(padded) | attn-out | qkv_wb | proj_wb
  __bf16* qws     = (__bf16*)d_ws;
  __bf16* kws     = qws + QSZ;
  __bf16* vws     = kws + QSZ;
  __bf16* ao      = vws + VSZ;
  __bf16* qkv_wb  = ao + AOSZ;
  __bf16* proj_wb = qkv_wb + W1SZ;

  {
    int n4 = (int)(W1SZ / 4);
    convert_w_kernel<<<dim3((n4 + 255) / 256), dim3(256), 0, stream>>>(
        qkv_w, qkv_wb, n4);
  }
  {
    int n4 = (int)(W2SZ / 4);
    convert_w_kernel<<<dim3((n4 + 255) / 256), dim3(256), 0, stream>>>(
        proj_w, proj_wb, n4);
  }
  swin_qkv_kernel<<<dim3(kBW), dim3(256), 0, stream>>>(x, qkv_wb, qkv_b, qws,
                                                       kws, vws);
  swin_attn_kernel<<<dim3(kBW * kNH), dim3(32), 0, stream>>>(qws, kws, vws,
                                                             rpb, ao);
  swin_proj_kernel<<<dim3(kBW), dim3(256), 0, stream>>>(ao, proj_wb, proj_b,
                                                        out);
}
